// GruAutoEncoder_6193342841056
// MI455X (gfx1250) — matrix operand order repaired and checked by reading, not compiled, not run on hardware
//
#include <hip/hip_runtime.h>
#include <cstdint>

// ---------------- types ----------------
typedef __bf16 bf16_t;
typedef __attribute__((ext_vector_type(16))) __bf16 v16bf;
typedef __attribute__((ext_vector_type(8)))  __bf16 v8bf;
typedef __attribute__((ext_vector_type(8)))  float  v8f;
typedef __attribute__((ext_vector_type(4))) unsigned int u32x4;
typedef __attribute__((ext_vector_type(8))) int          i32x8;
typedef __attribute__((ext_vector_type(4))) int          i32x4;

union V16U { v16bf v; v8bf h[2]; };

// ---------------- problem constants ----------------
#define BB 64
#define TT 512
#define EE 256
#define HH 256
#define GG 768          // 3H
#define VV 2000
#define TB (TT*BB)      // 32768
#define NEGF (-1e10f)
#define GXSLICE (16*GG) // 12288 bf16 elems = 24 KB per step per batch-group

// ---------------- WMMA helpers ----------------
__device__ __forceinline__ v8f wmma_bf16(v16bf a, v16bf b, v8f c) {
  return __builtin_amdgcn_wmma_f32_16x16x32_bf16(false, a, false, b, (short)0, c,
                                                 false, false);
}

// A tile (16 x 32, M x K): base = row 0 of tile, ldk = row stride (elements).
// lane m (0..15) <-> row m; half = lane/16 selects K interleave per ISA table.
__device__ __forceinline__ v16bf load_A_tile(const bf16_t* base, int ldk,
                                             int lane, int kbase) {
  int m = lane & 15, half = lane >> 4;
  V16U u;
  u.h[0] = *(const v8bf*)(base + (size_t)m * ldk + kbase + 8 * half);
  u.h[1] = *(const v8bf*)(base + (size_t)m * ldk + kbase + 16 + 8 * half);
  return u.v;
}

// B tile (32 x 16, K x N) taken from row-major weight W[N][K]:
// lane n holds column n of B = row n of W; K interleave per ISA table (8*half, 16+8*half).
__device__ __forceinline__ v16bf load_B_tile(const bf16_t* wbase, int ldk,
                                             int lane, int kbase) {
  int n = lane & 15, half = lane >> 4;
  V16U u;
  u.h[0] = *(const v8bf*)(wbase + (size_t)n * ldk + kbase + 8 * half);
  u.h[1] = *(const v8bf*)(wbase + (size_t)n * ldk + kbase + 16 + 8 * half);
  return u.v;
}

// ---------------- Tensor Data Mover: contiguous 24KB slice -> LDS ----------
// 1-D descriptor per CDNA5 ISA ch.8: count=1, data_size=2B,
// tensor_dim0 = tile_dim0 = GXSLICE, type=2 ("image").
// 6-arg builtin variant (clang-23 / therock headers): groups 2,3 + extra group
// are unused for a <=2D tensor and passed as zeros.
__device__ __forceinline__ void tdm_load_gx(unsigned lds_off, const bf16_t* gsrc) {
  unsigned long long ga = (unsigned long long)(uintptr_t)gsrc;
  u32x4 g0;
  g0[0] = 1u;                                               // count=1 (valid)
  g0[1] = lds_off;                                          // lds_addr (bytes)
  g0[2] = (unsigned)(ga & 0xffffffffu);                     // global_addr[31:0]
  g0[3] = (unsigned)((ga >> 32) & 0x01ffffffu) | (2u << 30);// addr[56:32]|type=2
  i32x8 g1;
  g1[0] = 0x00010000;                   // workgroup_mask=0, data_size=1 (2B)
  g1[1] = (int)((unsigned)GXSLICE << 16); // tensor_dim0[15:0] @ bits 63:48
  g1[2] = (int)(1u << 16);              // tensor_dim0[31:16]=0, tensor_dim1=1
  g1[3] = (int)((unsigned)GXSLICE << 16); // tile_dim0 @ bits 127:112
  g1[4] = 0;                            // tile_dim1=0, tile_dim2=0 (unused)
  g1[5] = GXSLICE;                      // tensor_dim0_stride[31:0]
  g1[6] = 0;
  g1[7] = 0;
  i32x4 gz4 = {0, 0, 0, 0};
  i32x8 gz8 = {0, 0, 0, 0, 0, 0, 0, 0};
  __builtin_amdgcn_tensor_load_to_lds(g0, g1, gz4, gz4, gz8, 0);
}

// ---------------- small utility kernels ----------------
__global__ void k_cvt_f32_bf16(const float* __restrict__ s, bf16_t* __restrict__ d,
                               long n) {
  long i = (long)blockIdx.x * blockDim.x + threadIdx.x;
  if (i < n) d[i] = (bf16_t)s[i];
}

// X[t][b][e] = emb[idx[b][t]][e]   (time-major, bf16)
__global__ void k_embed(const float* __restrict__ emb, const int* __restrict__ idx,
                        bf16_t* __restrict__ X) {
  long i = (long)blockIdx.x * blockDim.x + threadIdx.x;
  long n = (long)TT * BB * EE;
  if (i >= n) return;
  int e = (int)(i % EE);
  int b = (int)((i / EE) % BB);
  int t = (int)(i / ((long)EE * BB));
  X[i] = (bf16_t)emb[(long)idx[b * TT + t] * EE + e];
}

// EncOut[b][t][h] = Y1f[t][b][h] + Y1b[t][b][h]   (bf16 out, f32 add)
__global__ void k_encout(const bf16_t* __restrict__ yf, const bf16_t* __restrict__ yb,
                         bf16_t* __restrict__ enc) {
  long i = (long)blockIdx.x * blockDim.x + threadIdx.x;
  long n = (long)BB * TT * HH;
  if (i >= n) return;
  int h = (int)(i % HH);
  int t = (int)((i / HH) % TT);
  int b = (int)(i / ((long)HH * TT));
  long src = ((long)t * BB + b) * HH + h;
  enc[i] = (bf16_t)((float)yf[src] + (float)yb[src]);
}

__global__ void k_zero2(float* a) { a[0] = 0.f; a[1] = 0.f; }

__global__ void k_finalize(const float* __restrict__ acc, float* __restrict__ out) {
  out[0] = acc[0] / fmaxf(acc[1], 1.0f);
}

// ---------------- generic bf16 GEMM: C[M,N] = A[M,K] @ W[N,K]^T (+bias) ----------
template <bool OUTBF16>
__global__ void k_gemm(const bf16_t* __restrict__ A, const bf16_t* __restrict__ W,
                       float* __restrict__ Cf, bf16_t* __restrict__ Cb,
                       const float* __restrict__ bias, int M, int N, int K) {
  int wv = threadIdx.x >> 5, ln = threadIdx.x & 31;
  int tilesN = N >> 4;
  long tiles = (long)(M >> 4) * tilesN;
  long tile = (long)blockIdx.x * 8 + wv;
  if (tile >= tiles) return;
  int tm = (int)(tile / tilesN), tn = (int)(tile % tilesN);
  const bf16_t* Ab = A + (long)tm * 16 * K;
  const bf16_t* Wb = W + (long)tn * 16 * K;
  v8f acc = {};
  for (int kc = 0; kc < K; kc += 32) {
    v16bf a = load_A_tile(Ab, K, ln, kc);
    v16bf b = load_B_tile(Wb, K, ln, kc);
    acc = wmma_bf16(a, b, acc);
  }
  int col = tn * 16 + (ln & 15);
  int mrow = tm * 16 + 8 * (ln >> 4);
  float bz = bias ? bias[col] : 0.f;
#pragma unroll
  for (int r = 0; r < 8; ++r) {
    float v = acc[r] + bz;
    long idx = (long)(mrow + r) * N + col;
    if (OUTBF16) Cb[idx] = (bf16_t)v; else Cf[idx] = v;
  }
}

// ---------------- encoder GRU scan ----------------
// grid.x = 8: blockIdx >> 2 = direction, blockIdx & 3 = batch group (16 rows).
// 512 threads = 16 waves. Per step: TDM-prefetch gx[t+1] into LDS, gh = h@Whh^T
// via WMMA, then gate math from LDS-resident gx.
__global__ __launch_bounds__(512)
void k_gru_scan(const bf16_t* __restrict__ gx_f, const bf16_t* __restrict__ gx_b,
                const bf16_t* __restrict__ whh_f, const bf16_t* __restrict__ whh_b,
                const float* __restrict__ bih_f, const float* __restrict__ bih_b,
                const float* __restrict__ bhh_f, const float* __restrict__ bhh_b,
                bf16_t* __restrict__ yf, bf16_t* __restrict__ yb, int ystrideB,
                bf16_t* __restrict__ hfin) {
  __shared__ float  s_gh[16 * GG];            // 48 KB
  __shared__ bf16_t s_h[16 * HH];             // 8 KB
  __shared__ bf16_t s_gx[2 * GXSLICE];        // 48 KB double buffer
  int dir = blockIdx.x >> 2;
  int b0 = (blockIdx.x & 3) * 16;
  const bf16_t* gx  = dir ? gx_b  : gx_f;
  const bf16_t* whh = dir ? whh_b : whh_f;
  const float*  bih = dir ? bih_b : bih_f;
  const float*  bhh = dir ? bhh_b : bhh_f;
  bf16_t*       yo  = dir ? yb    : yf;

  for (int i = threadIdx.x; i < 16 * HH; i += 512) s_h[i] = (bf16_t)0.0f;

  int wv = threadIdx.x >> 5, ln = threadIdx.x & 31;
  unsigned gx_lds0 = (unsigned)(uintptr_t)(void*)&s_gx[0];
  unsigned gx_lds1 = (unsigned)(uintptr_t)(void*)&s_gx[GXSLICE];

  // prefetch step 0
  if (wv == 0) {
    int t0 = dir ? (TT - 1) : 0;
    tdm_load_gx(gx_lds0, gx + ((long)t0 * BB + b0) * GG);
  }

  for (int s = 0; s < TT; ++s) {
    int t = dir ? (TT - 1 - s) : s;
    __syncthreads();
    // issue DMA for next step's gx slice (overlaps with WMMA below)
    if (wv == 0 && s + 1 < TT) {
      int tn = dir ? (TT - 2 - s) : (s + 1);
      tdm_load_gx((s & 1) ? gx_lds0 : gx_lds1, gx + ((long)tn * BB + b0) * GG);
    }
    // gh[16][768] = h[16][256] @ whh[768][256]^T : 48 N-tiles, 3 per wave
#pragma unroll
    for (int tt = 0; tt < 3; ++tt) {
      int nt = wv + tt * 16;
      const bf16_t* wb = whh + (long)nt * 16 * HH;
      v8f acc = {};
#pragma unroll
      for (int kc = 0; kc < 8; ++kc) {
        v16bf a = load_A_tile(s_h, HH, ln, kc * 32);
        v16bf b = load_B_tile(wb, HH, ln, kc * 32);
        acc = wmma_bf16(a, b, acc);
      }
      int col = nt * 16 + (ln & 15);
      int mb = 8 * (ln >> 4);
#pragma unroll
      for (int r = 0; r < 8; ++r) s_gh[(mb + r) * GG + col] = acc[r];
    }
    // wave 0 guarantees this step's gx DMA has landed before the barrier
    if (wv == 0) {
      if (s + 1 < TT) __builtin_amdgcn_s_wait_tensorcnt(1);
      else            __builtin_amdgcn_s_wait_tensorcnt(0);
    }
    __syncthreads();
    // gate math: 16*256 elements, 8 per thread; gx comes from LDS
    const bf16_t* gsl = &s_gx[(s & 1) * GXSLICE];
    for (int e = threadIdx.x; e < 16 * HH; e += 512) {
      int m = e >> 8, j = e & 255;
      int b = b0 + m;
      const bf16_t* g = gsl + m * GG;
      float xr = (float)g[j]        + bih[j];
      float xz = (float)g[HH + j]   + bih[HH + j];
      float xn = (float)g[2*HH + j] + bih[2*HH + j];
      float hr = s_gh[m * GG + j]        + bhh[j];
      float hz = s_gh[m * GG + HH + j]   + bhh[HH + j];
      float hn = s_gh[m * GG + 2*HH + j] + bhh[2*HH + j];
      float r = 1.f / (1.f + __expf(-(xr + hr)));
      float z = 1.f / (1.f + __expf(-(xz + hz)));
      float nn = tanhf(xn + r * hn);
      float hold = (float)s_h[e];
      float hnew = (1.f - z) * nn + z * hold;
      s_h[e] = (bf16_t)hnew;
      yo[(long)t * BB * ystrideB + (long)b * ystrideB + j] = (bf16_t)hnew;
    }
  }
  __syncthreads();
  if (hfin) {
    for (int e = threadIdx.x; e < 16 * HH; e += 512) {
      int m = e >> 8, j = e & 255;
      hfin[(long)(b0 + m) * (2 * HH) + dir * HH + j] = s_h[e];
    }
  }
}

// ---------------- decoder scan: GRU + dot attention + concat ----------------
// grid.x = 4 batch groups of 16 rows, 512 threads = 16 waves.
__global__ __launch_bounds__(512)
void k_dec_scan(const bf16_t* __restrict__ gx,      // [T][B][768]
                const bf16_t* __restrict__ whh,     // [768][256]
                const float* __restrict__ bih, const float* __restrict__ bhh,
                const float* __restrict__ hinit,    // [B][256] f32
                const bf16_t* __restrict__ enc,     // [B][T][256]
                const int* __restrict__ mask,       // [B][T]
                const bf16_t* __restrict__ wcat,    // [256][512]
                const float* __restrict__ bcat,     // [256]
                bf16_t* __restrict__ CC)            // [T][B][256]
{
  __shared__ float  s_gh[16 * GG];        // 48 KB
  __shared__ float  s_att[16 * TT];       // 32 KB
  __shared__ bf16_t s_cat[16 * 512];      // 16 KB: cols [0,256)=h, [256,512)=ctx
  __shared__ bf16_t s_gx[2 * GXSLICE];    // 48 KB double buffer
  int b0 = blockIdx.x * 16;
  int wv = threadIdx.x >> 5, ln = threadIdx.x & 31;
  unsigned gx_lds0 = (unsigned)(uintptr_t)(void*)&s_gx[0];
  unsigned gx_lds1 = (unsigned)(uintptr_t)(void*)&s_gx[GXSLICE];

  for (int e = threadIdx.x; e < 16 * HH; e += 512) {
    int m = e >> 8, j = e & 255;
    s_cat[m * 512 + j] = (bf16_t)hinit[(long)(b0 + m) * HH + j];
  }
  if (wv == 0) tdm_load_gx(gx_lds0, gx + (long)b0 * GG);

  for (int t = 0; t < TT; ++t) {
    __syncthreads();
    if (wv == 0 && t + 1 < TT)
      tdm_load_gx((t & 1) ? gx_lds0 : gx_lds1, gx + ((long)(t + 1) * BB + b0) * GG);
    // ---- gh = h @ whh^T ----
#pragma unroll
    for (int tt = 0; tt < 3; ++tt) {
      int nt = wv + tt * 16;
      const bf16_t* wb = whh + (long)nt * 16 * HH;
      v8f acc = {};
#pragma unroll
      for (int kc = 0; kc < 8; ++kc) {
        v16bf a = load_A_tile(s_cat, 512, ln, kc * 32);
        v16bf b = load_B_tile(wb, HH, ln, kc * 32);
        acc = wmma_bf16(a, b, acc);
      }
      int col = nt * 16 + (ln & 15);
      int mb = 8 * (ln >> 4);
#pragma unroll
      for (int r = 0; r < 8; ++r) s_gh[(mb + r) * GG + col] = acc[r];
    }
    if (wv == 0) {
      if (t + 1 < TT) __builtin_amdgcn_s_wait_tensorcnt(1);
      else            __builtin_amdgcn_s_wait_tensorcnt(0);
    }
    __syncthreads();
    // ---- gates (gx from LDS) ----
    const bf16_t* gsl = &s_gx[(t & 1) * GXSLICE];
    for (int e = threadIdx.x; e < 16 * HH; e += 512) {
      int m = e >> 8, j = e & 255;
      const bf16_t* g = gsl + m * GG;
      float xr = (float)g[j]        + bih[j];
      float xz = (float)g[HH + j]   + bih[HH + j];
      float xn = (float)g[2*HH + j] + bih[2*HH + j];
      float hr = s_gh[m * GG + j]        + bhh[j];
      float hz = s_gh[m * GG + HH + j]   + bhh[HH + j];
      float hn = s_gh[m * GG + 2*HH + j] + bhh[2*HH + j];
      float r = 1.f / (1.f + __expf(-(xr + hr)));
      float z = 1.f / (1.f + __expf(-(xz + hz)));
      float nn = tanhf(xn + r * hn);
      float hold = (float)s_cat[m * 512 + j];
      s_cat[m * 512 + j] = (bf16_t)((1.f - z) * nn + z * hold);
    }
    __syncthreads();
    // ---- energies: att[m][tp] = h[m] . enc[b][tp] (masked) ----
    for (int e = threadIdx.x; e < 16 * TT; e += 512) {
      int m = e >> 9, tp = e & 511;
      int b = b0 + m;
      const bf16_t* ev = enc + ((long)b * TT + tp) * HH;
      float acc = 0.f;
#pragma unroll 4
      for (int j = 0; j < HH; ++j) acc += (float)s_cat[m * 512 + j] * (float)ev[j];
      s_att[m * TT + tp] = (mask[b * TT + tp] == 0) ? NEGF : acc;
    }
    __syncthreads();
    // ---- softmax: wave wv handles row wv ----
    {
      float v[16];
      float mx = -__builtin_inff();
#pragma unroll
      for (int k = 0; k < 16; ++k) {
        v[k] = s_att[wv * TT + ln + k * 32];
        mx = fmaxf(mx, v[k]);
      }
      for (int off = 16; off > 0; off >>= 1) mx = fmaxf(mx, __shfl_xor(mx, off, 32));
      float sm = 0.f;
#pragma unroll
      for (int k = 0; k < 16; ++k) { v[k] = __expf(v[k] - mx); sm += v[k]; }
      for (int off = 16; off > 0; off >>= 1) sm += __shfl_xor(sm, off, 32);
      float inv = 1.f / sm;
#pragma unroll
      for (int k = 0; k < 16; ++k) s_att[wv * TT + ln + k * 32] = v[k] * inv;
    }
    __syncthreads();
    // ---- ctx[m][j] = sum_tp att * enc ----
    for (int e = threadIdx.x; e < 16 * HH; e += 512) {
      int m = e >> 8, j = e & 255;
      int b = b0 + m;
      float acc = 0.f;
#pragma unroll 4
      for (int tp = 0; tp < TT; ++tp)
        acc += s_att[m * TT + tp] * (float)enc[((long)b * TT + tp) * HH + j];
      s_cat[m * 512 + HH + j] = (bf16_t)acc;
    }
    __syncthreads();
    // ---- cc = tanh(cat @ wcat^T + bcat): 16 N-tiles, one per wave ----
    {
      int nt = wv;
      const bf16_t* wb = wcat + (long)nt * 16 * 512;
      v8f acc = {};
#pragma unroll
      for (int kc = 0; kc < 16; ++kc) {
        v16bf a = load_A_tile(s_cat, 512, ln, kc * 32);
        v16bf b = load_B_tile(wb, 512, ln, kc * 32);
        acc = wmma_bf16(a, b, acc);
      }
      int col = nt * 16 + (ln & 15);
      int mb = 8 * (ln >> 4);
#pragma unroll
      for (int r = 0; r < 8; ++r) {
        float v = tanhf(acc[r] + bcat[col]);
        CC[((long)t * BB + b0 + mb + r) * HH + col] = (bf16_t)v;
      }
    }
  }
}

// ---------------- masked cross-entropy over logits rows ----------------
__global__ void k_loss(const float* __restrict__ logits, const int* __restrict__ tgt,
                       float* __restrict__ acc) {
  __shared__ float red[256];
  long row = blockIdx.x;
  const float* lg = logits + row * VV;
  int tg = tgt[row];
  float mx = -__builtin_inff();
  for (int j = threadIdx.x; j < VV; j += 256) mx = fmaxf(mx, lg[j]);
  red[threadIdx.x] = mx; __syncthreads();
  for (int s = 128; s > 0; s >>= 1) {
    if (threadIdx.x < s) red[threadIdx.x] = fmaxf(red[threadIdx.x], red[threadIdx.x + s]);
    __syncthreads();
  }
  mx = red[0]; __syncthreads();
  float sm = 0.f;
  for (int j = threadIdx.x; j < VV; j += 256) sm += __expf(lg[j] - mx);
  red[threadIdx.x] = sm; __syncthreads();
  for (int s = 128; s > 0; s >>= 1) {
    if (threadIdx.x < s) red[threadIdx.x] += red[threadIdx.x + s];
    __syncthreads();
  }
  if (threadIdx.x == 0 && tg != 0) {
    float nll = -(lg[tg] - mx - __logf(red[0]));
    atomicAdd(acc, nll);
    atomicAdd(acc + 1, 1.0f);
  }
}

// ---------------- host orchestration ----------------
extern "C" void kernel_launch(void* const* d_in, const int* in_sizes, int n_in,
                              void* d_out, int out_size, void* d_ws, size_t ws_size,
                              hipStream_t stream) {
  const int*   mask     = (const int*)d_in[0];
  const int*   enc_inp  = (const int*)d_in[1];
  const int*   dec_inp  = (const int*)d_in[2];
  const float* emb_enc  = (const float*)d_in[3];
  const float* emb_dec  = (const float*)d_in[4];
  const float* eWih0    = (const float*)d_in[5];
  const float* eWhh0    = (const float*)d_in[6];
  const float* ebih0    = (const float*)d_in[7];
  const float* ebhh0    = (const float*)d_in[8];
  const float* eWih1    = (const float*)d_in[9];
  const float* eWhh1    = (const float*)d_in[10];
  const float* ebih1    = (const float*)d_in[11];
  const float* ebhh1    = (const float*)d_in[12];
  const float* fcW      = (const float*)d_in[13];
  const float* fcb      = (const float*)d_in[14];
  const float* dWih     = (const float*)d_in[15];
  const float* dWhh     = (const float*)d_in[16];
  const float* dbih     = (const float*)d_in[17];
  const float* dbhh     = (const float*)d_in[18];
  const float* catW     = (const float*)d_in[19];
  const float* catb     = (const float*)d_in[20];
  const float* outW     = (const float*)d_in[21];
  const float* outb     = (const float*)d_in[22];

  // workspace carve-out
  char* p = (char*)d_ws;
  size_t off = 0;
  auto take = [&](size_t bytes) { char* r = p + off; off = (off + bytes + 255) & ~(size_t)255; return r; };
  bf16_t* w_eWih0 = (bf16_t*)take(2L*GG*EE*2);
  bf16_t* w_eWhh0 = (bf16_t*)take(2L*GG*HH*2);
  bf16_t* w_eWih1 = (bf16_t*)take(2L*GG*512*2);
  bf16_t* w_eWhh1 = (bf16_t*)take(2L*GG*HH*2);
  bf16_t* w_dWih  = (bf16_t*)take((long)GG*EE*2);
  bf16_t* w_dWhh  = (bf16_t*)take((long)GG*HH*2);
  bf16_t* w_fc    = (bf16_t*)take((long)HH*512*2);
  bf16_t* w_cat   = (bf16_t*)take((long)HH*512*2);
  bf16_t* w_out   = (bf16_t*)take((long)VV*HH*2);
  bf16_t* Xenc    = (bf16_t*)take((long)TB*EE*2);
  bf16_t* Xdec    = (bf16_t*)take((long)TB*EE*2);
  bf16_t* Y0      = (bf16_t*)take((long)TB*512*2);
  bf16_t* Y1f     = (bf16_t*)take((long)TB*HH*2);
  bf16_t* Y1b     = (bf16_t*)take((long)TB*HH*2);
  bf16_t* EncOut  = (bf16_t*)take((long)BB*TT*HH*2);
  bf16_t* Hcat    = (bf16_t*)take((long)BB*512*2);
  float*  Hinit   = (float*) take((long)BB*HH*4);
  bf16_t* CC      = (bf16_t*)take((long)TB*HH*2);
  float*  acc     = (float*) take(256);

  // gate-precompute buffers live in d_out (overwritten by final logits GEMM)
  bf16_t* GXf = (bf16_t*)d_out;
  bf16_t* GXb = (bf16_t*)d_out + (long)TB * GG;
  float*  logits = (float*)d_out;

  const int THR = 256;
  auto g1 = [&](long n) { return dim3((unsigned)((n + THR - 1) / THR)); };

  // 1) weights -> bf16
  k_cvt_f32_bf16<<<g1(2L*GG*EE), THR, 0, stream>>>(eWih0, w_eWih0, 2L*GG*EE);
  k_cvt_f32_bf16<<<g1(2L*GG*HH), THR, 0, stream>>>(eWhh0, w_eWhh0, 2L*GG*HH);
  k_cvt_f32_bf16<<<g1(2L*GG*512), THR, 0, stream>>>(eWih1, w_eWih1, 2L*GG*512);
  k_cvt_f32_bf16<<<g1(2L*GG*HH), THR, 0, stream>>>(eWhh1, w_eWhh1, 2L*GG*HH);
  k_cvt_f32_bf16<<<g1((long)GG*EE), THR, 0, stream>>>(dWih, w_dWih, (long)GG*EE);
  k_cvt_f32_bf16<<<g1((long)GG*HH), THR, 0, stream>>>(dWhh, w_dWhh, (long)GG*HH);
  k_cvt_f32_bf16<<<g1((long)HH*512), THR, 0, stream>>>(fcW, w_fc, (long)HH*512);
  k_cvt_f32_bf16<<<g1((long)HH*512), THR, 0, stream>>>(catW, w_cat, (long)HH*512);
  k_cvt_f32_bf16<<<g1((long)VV*HH), THR, 0, stream>>>(outW, w_out, (long)VV*HH);

  // 2) embeddings (time-major bf16)
  k_embed<<<g1((long)TB*EE), THR, 0, stream>>>(emb_enc, enc_inp, Xenc);
  k_embed<<<g1((long)TB*EE), THR, 0, stream>>>(emb_dec, dec_inp, Xdec);

  auto gemm_blocks = [&](int M, int N) { return dim3((unsigned)(((long)(M/16)*(N/16) + 7) / 8)); };

  // 3) encoder layer 0: input gates (both dirs), then scan
  k_gemm<true><<<gemm_blocks(TB, GG), THR, 0, stream>>>(Xenc, w_eWih0, nullptr, GXf, nullptr, TB, GG, EE);
  k_gemm<true><<<gemm_blocks(TB, GG), THR, 0, stream>>>(Xenc, w_eWih0 + (long)GG*EE, nullptr, GXb, nullptr, TB, GG, EE);
  k_gru_scan<<<8, 512, 0, stream>>>(GXf, GXb,
      w_eWhh0, w_eWhh0 + (long)GG*HH, ebih0, ebih0 + GG, ebhh0, ebhh0 + GG,
      Y0, Y0 + HH, 512, nullptr);

  // 4) encoder layer 1
  k_gemm<true><<<gemm_blocks(TB, GG), THR, 0, stream>>>(Y0, w_eWih1, nullptr, GXf, nullptr, TB, GG, 512);
  k_gemm<true><<<gemm_blocks(TB, GG), THR, 0, stream>>>(Y0, w_eWih1 + (long)GG*512, nullptr, GXb, nullptr, TB, GG, 512);
  k_gru_scan<<<8, 512, 0, stream>>>(GXf, GXb,
      w_eWhh1, w_eWhh1 + (long)GG*HH, ebih1, ebih1 + GG, ebhh1, ebhh1 + GG,
      Y1f, Y1b, HH, Hcat);

  // 5) enc_out = sum of dirs, transposed to [B,T,H]; decoder init hidden
  k_encout<<<g1((long)BB*TT*HH), THR, 0, stream>>>(Y1f, Y1b, EncOut);
  k_gemm<false><<<gemm_blocks(BB, HH), THR, 0, stream>>>(Hcat, w_fc, Hinit, nullptr, fcb, BB, HH, 512);

  // 6) decoder: input gates then scan (GRU + attention + concat)
  k_gemm<true><<<gemm_blocks(TB, GG), THR, 0, stream>>>(Xdec, w_dWih, nullptr, GXf, nullptr, TB, GG, EE);
  k_dec_scan<<<4, 512, 0, stream>>>(GXf, w_dWhh, dbih, dbhh, Hinit, EncOut, mask,
                                    w_cat, catb, CC);

  // 7) logits = CC @ outW^T + outb  -> d_out [T,B,V]
  k_gemm<false><<<gemm_blocks(TB, VV), THR, 0, stream>>>(CC, w_out, logits, nullptr, outb, TB, VV, HH);

  // 8) masked cross-entropy -> d_out[T*B*V]
  k_zero2<<<1, 1, 0, stream>>>(acc);
  k_loss<<<TB, 256, 0, stream>>>(logits, enc_inp, acc);
  k_finalize<<<1, 1, 0, stream>>>(acc, logits + (long)TB * VV);
}
